// ExBimamba_75883482186472
// MI455X (gfx1250) — matrix operand order repaired and checked by reading, not compiled, not run on hardware
//
#include <hip/hip_runtime.h>
#include <hip/hip_bf16.h>
#include <math.h>

#define D_MODEL 512
#define D_STATE 16
#define D_CONV  4
#define D_IN    1024
#define DT_RANK 32
#define BB      4
#define LL      1024
#define BL      (BB*LL)   // 4096 rows
#define NCHUNK  16
#define CLEN    (LL/NCHUNK)   // 64

typedef __attribute__((ext_vector_type(16))) __bf16 v16bf;
typedef __attribute__((ext_vector_type(8)))  __bf16 v8bf;
typedef __attribute__((ext_vector_type(8)))  float  v8f;

__device__ __forceinline__ float sigmoidf_(float x){ return 1.0f/(1.0f+__expf(-x)); }
__device__ __forceinline__ float siluf_(float x){ return x*sigmoidf_(x); }
__device__ __forceinline__ float softplusf_(float x){ return (x>20.f)?x:log1pf(__expf(x)); }

// ---------------- lengths = mask.sum(axis=1) ----------------
__global__ void k_lengths(const int* __restrict__ mask, int* __restrict__ lengths){
  __shared__ int sh[256];
  int b = blockIdx.x, t = threadIdx.x;
  int s = 0;
  for (int i=t;i<LL;i+=256) s += mask[b*LL+i];
  sh[t]=s; __syncthreads();
  for (int k=128;k>0;k>>=1){ if(t<k) sh[t]+=sh[t+k]; __syncthreads(); }
  if (t==0) lengths[b]=sh[0];
}

// ---------------- cast x -> bf16 (normal + masked-flip) ----------------
__global__ void k_castx(const float* __restrict__ x, const int* __restrict__ lengths,
                        __bf16* __restrict__ xbf, __bf16* __restrict__ xbfr){
  int idx = blockIdx.x*blockDim.x + threadIdx.x;
  if (idx >= BL*D_MODEL) return;
  int row = idx / D_MODEL, d = idx % D_MODEL;
  int b = row / LL, l = row % LL;
  xbf[idx] = (__bf16)x[idx];
  int len = lengths[b];
  int lf = (l < len) ? (len-1-l) : l;
  xbfr[idx] = (__bf16)x[((size_t)(b*LL+lf))*D_MODEL + d];
}

__global__ void k_cast(const float* __restrict__ src, __bf16* __restrict__ dst, int n){
  int i = blockIdx.x*blockDim.x + threadIdx.x;
  if (i<n) dst[i] = (__bf16)src[i];
}

__global__ void k_cast_cols(const float* __restrict__ src, __bf16* __restrict__ dst,
                            int rows, int src_ld, int cols){
  int i = blockIdx.x*blockDim.x + threadIdx.x;
  if (i >= rows*cols) return;
  int r = i/cols, c = i%cols;
  dst[i] = (__bf16)src[(size_t)r*src_ld + c];
}

// ------- WMMA bf16 GEMM, 2x2 register-blocked: C(MxN) = A(MxK) @ Bw(NxK)^T -------
// one wave computes a 32x32 output tile (4 accumulators); blockDim (32,2) -> 64 cols/block
__device__ __forceinline__ v16bf load_afrag(const __bf16* arow, int k0, int h){
  v8bf lo = *(const v8bf*)(arow + k0 + 16*h);
  v8bf hi = *(const v8bf*)(arow + k0 + 16*h + 8);
  v16bf af;
  #pragma unroll
  for (int i=0;i<8;++i){ af[i]=lo[i]; af[i+8]=hi[i]; }
  return af;
}

__global__ void k_gemm_wmma(const __bf16* __restrict__ A, const __bf16* __restrict__ Bw,
                            const float* __restrict__ bias,
                            float* __restrict__ Cf, __bf16* __restrict__ Cbf,
                            int M, int N, int K, int act){
  const int lane = threadIdx.x;
  const int h    = lane >> 4;               // half-wave selector
  const int m    = lane & 15;
  const int nblk = blockIdx.x*2 + threadIdx.y;  // 32-column block
  const int mblk = blockIdx.y;                  // 32-row block
  const int col0 = nblk*32 + m, col1 = col0 + 16;
  const __bf16* a0 = A  + (size_t)(mblk*32 + m)*K;
  const __bf16* a1 = A  + (size_t)(mblk*32 + 16 + m)*K;
  const __bf16* b0 = Bw + (size_t)col0*K + 16*h;
  const __bf16* b1 = Bw + (size_t)col1*K + 16*h;
  v8f acc00 = {0.f,0.f,0.f,0.f,0.f,0.f,0.f,0.f};
  v8f acc01 = acc00, acc10 = acc00, acc11 = acc00;
  for (int k0=0; k0<K; k0+=32){
    v16bf af0 = load_afrag(a0, k0, h);
    v16bf af1 = load_afrag(a1, k0, h);
    v16bf bf0 = *(const v16bf*)(b0 + k0);
    v16bf bf1 = *(const v16bf*)(b1 + k0);
    __builtin_prefetch(a0 + k0 + 128, 0, 1);
    __builtin_prefetch(b0 + k0 + 128, 0, 1);
    acc00 = __builtin_amdgcn_wmma_f32_16x16x32_bf16(false, af0, false, bf0, (short)0, acc00, false, false);
    acc01 = __builtin_amdgcn_wmma_f32_16x16x32_bf16(false, af0, false, bf1, (short)0, acc01, false, false);
    acc10 = __builtin_amdgcn_wmma_f32_16x16x32_bf16(false, af1, false, bf0, (short)0, acc10, false, false);
    acc11 = __builtin_amdgcn_wmma_f32_16x16x32_bf16(false, af1, false, bf1, (short)0, acc11, false, false);
  }
  float bv0 = bias ? bias[col0] : 0.0f;
  float bv1 = bias ? bias[col1] : 0.0f;
  #pragma unroll
  for (int ti=0; ti<2; ++ti){
    v8f accA = ti ? acc10 : acc00;
    v8f accB = ti ? acc11 : acc01;
    #pragma unroll
    for (int r=0;r<8;++r){                  // VGPR r -> row r+8h within 16x16 tile
      int row = mblk*32 + ti*16 + r + 8*h;
      float v0 = accA[r] + bv0;
      float v1 = accB[r] + bv1;
      if (act==1){ v0 = softplusf_(v0); v1 = softplusf_(v1); }
      size_t o0 = (size_t)row*N + col0;
      size_t o1 = (size_t)row*N + col1;
      if (Cf) { Cf[o0]  = v0;          Cf[o1]  = v1; }
      if (Cbf){ Cbf[o0] = (__bf16)v0;  Cbf[o1] = (__bf16)v1; }
    }
  }
}

// ---------------- causal depthwise conv(4) + bias + SiLU ----------------
__global__ void k_conv_silu(const float* __restrict__ xz, const float* __restrict__ cw,
                            const float* __restrict__ cb,
                            float* __restrict__ xhf, __bf16* __restrict__ xhbf){
  int idx = blockIdx.x*blockDim.x + threadIdx.x;
  if (idx >= BL*D_IN) return;
  int row = idx / D_IN, c = idx % D_IN;
  int b = row / LL, l = row % LL;
  float acc = cb[c];
  #pragma unroll
  for (int t=0;t<D_CONV;++t){
    int ls = l - (D_CONV-1) + t;
    if (ls >= 0)
      acc = fmaf(cw[c*D_CONV + t], xz[((size_t)(b*LL+ls))*(2*D_IN) + c], acc);
  }
  float v = siluf_(acc);
  xhf[idx]  = v;
  xhbf[idx] = (__bf16)v;
}

// ======== chunked parallel selective scan (associative recurrence) ========
// lane <-> (b, chunk, d): idx = chunk*D_IN + d so VMEM stays coalesced over d.
// phase1: per-chunk transition  h_out = P*h_in + q   (P = prod dA, q = local h)
__global__ void k_scan_phase1(const float* __restrict__ delta, const float* __restrict__ xhf,
                              const float* __restrict__ xdbl, const float* __restrict__ A_log,
                              float* __restrict__ Pbuf, float* __restrict__ qbuf){
  int idx = blockIdx.x*blockDim.x + threadIdx.x;   // 0 .. NCHUNK*D_IN-1
  int b = blockIdx.y;
  int d = idx % D_IN, chunk = idx / D_IN;
  float a[D_STATE], hst[D_STATE], P[D_STATE];
  #pragma unroll
  for (int n=0;n<D_STATE;++n){ a[n] = -__expf(A_log[d*D_STATE+n]); hst[n]=0.f; P[n]=1.f; }
  int l0 = chunk*CLEN;
  for (int l=l0; l<l0+CLEN; ++l){
    size_t row = (size_t)b*LL + l;
    float dl = delta[row*D_IN + d];
    float dx = dl * xhf[row*D_IN + d];
    const float* bc = xdbl + row*(DT_RANK+2*D_STATE);
    #pragma unroll
    for (int n=0;n<D_STATE;++n){
      float dA = __expf(dl * a[n]);
      hst[n] = fmaf(dA, hst[n], dx * bc[DT_RANK + n]);
      P[n] *= dA;
    }
  }
  size_t o = (((size_t)b*NCHUNK + chunk)*D_IN + d)*D_STATE;
  #pragma unroll
  for (int n=0;n<D_STATE;++n){ Pbuf[o+n] = P[n]; qbuf[o+n] = hst[n]; }
}

// phase2: sequential combine over chunks; one lane per (b,d,n)
__global__ void k_scan_phase2(const float* __restrict__ Pbuf, const float* __restrict__ qbuf,
                              float* __restrict__ hin){
  int idx = blockIdx.x*blockDim.x + threadIdx.x;   // 0 .. BB*D_IN*D_STATE-1
  if (idx >= BB*D_IN*D_STATE) return;
  int n = idx % D_STATE;
  int d = (idx / D_STATE) % D_IN;
  int b = idx / (D_STATE*D_IN);
  float h = 0.f;
  for (int c=0; c<NCHUNK; ++c){
    size_t o = (((size_t)b*NCHUNK + c)*D_IN + d)*D_STATE + n;
    hin[o] = h;
    h = fmaf(Pbuf[o], h, qbuf[o]);
  }
}

// phase3: re-run each chunk seeded with incoming state, emit ys
__global__ void k_scan_phase3(const float* __restrict__ delta, const float* __restrict__ xhf,
                              const float* __restrict__ xdbl, const float* __restrict__ A_log,
                              const float* __restrict__ hin, float* __restrict__ ys){
  int idx = blockIdx.x*blockDim.x + threadIdx.x;
  int b = blockIdx.y;
  int d = idx % D_IN, chunk = idx / D_IN;
  size_t o = (((size_t)b*NCHUNK + chunk)*D_IN + d)*D_STATE;
  float a[D_STATE], hst[D_STATE];
  #pragma unroll
  for (int n=0;n<D_STATE;++n){ a[n] = -__expf(A_log[d*D_STATE+n]); hst[n]=hin[o+n]; }
  int l0 = chunk*CLEN;
  for (int l=l0; l<l0+CLEN; ++l){
    size_t row = (size_t)b*LL + l;
    float dl = delta[row*D_IN + d];
    float dx = dl * xhf[row*D_IN + d];
    const float* bc = xdbl + row*(DT_RANK+2*D_STATE);
    float accv = 0.f;
    #pragma unroll
    for (int n=0;n<D_STATE;++n){
      float dA = __expf(dl * a[n]);
      hst[n] = fmaf(dA, hst[n], dx * bc[DT_RANK + n]);
      accv   = fmaf(hst[n], bc[DT_RANK + D_STATE + n], accv);
    }
    ys[row*D_IN + d] = accv;
  }
}

// ---------------- y = (ys + xh*D) * silu(z) -> bf16 ----------------
__global__ void k_gate(const float* __restrict__ ys, const float* __restrict__ xhf,
                       const float* __restrict__ xz, const float* __restrict__ Dp,
                       __bf16* __restrict__ ybf){
  int idx = blockIdx.x*blockDim.x + threadIdx.x;
  if (idx >= BL*D_IN) return;
  int row = idx / D_IN, c = idx % D_IN;
  float z = xz[(size_t)row*(2*D_IN) + D_IN + c];
  float v = (ys[idx] + xhf[idx]*Dp[c]) * siluf_(z);
  ybf[idx] = (__bf16)v;
}

// ---------------- concat(fwd, masked-flip(bwd)) -> bf16 ----------------
__global__ void k_concat(const float* __restrict__ fwd, const float* __restrict__ bwd,
                         const int* __restrict__ lengths, __bf16* __restrict__ res){
  int idx = blockIdx.x*blockDim.x + threadIdx.x;
  if (idx >= BL*(2*D_MODEL)) return;
  int row = idx / (2*D_MODEL), e = idx % (2*D_MODEL);
  int b = row / LL, l = row % LL;
  float v;
  if (e < D_MODEL) v = fwd[(size_t)row*D_MODEL + e];
  else {
    int len = lengths[b];
    int lf = (l < len) ? (len-1-l) : l;
    v = bwd[((size_t)(b*LL+lf))*D_MODEL + (e - D_MODEL)];
  }
  res[idx] = (__bf16)v;
}

static inline void launch_gemm(hipStream_t s, const __bf16* A, const __bf16* Bw,
                               const float* bias, float* Cf, __bf16* Cbf,
                               int M,int N,int K,int act){
  dim3 blk(32,2), grd(N/64, M/32);
  k_gemm_wmma<<<grd, blk, 0, s>>>(A, Bw, bias, Cf, Cbf, M, N, K, act);
}

extern "C" void kernel_launch(void* const* d_in, const int* in_sizes, int n_in,
                              void* d_out, int out_size, void* d_ws, size_t ws_size,
                              hipStream_t stream){
  (void)in_sizes; (void)n_in; (void)out_size; (void)ws_size;
  const float* x    = (const float*)d_in[0];
  const int*   mask = (const int*)d_in[1];
  const float* Wo   = (const float*)d_in[2];
  const float* bo   = (const float*)d_in[3];
  const float* fP[9]; const float* bP[9];
  for (int i=0;i<9;++i){ fP[i]=(const float*)d_in[4+i]; bP[i]=(const float*)d_in[13+i]; }
  // P: 0 W_in, 1 conv_w, 2 conv_b, 3 W_x, 4 W_dt, 5 b_dt, 6 A_log, 7 D, 8 W_out

  char* ws = (char*)d_ws; size_t off = 0;
  auto alloc = [&](size_t bytes)->char*{
    char* p = ws + off; off += (bytes + 255) & ~(size_t)255; return p; };
  int*    lengths = (int*)   alloc(BB*sizeof(int));
  __bf16* xbf     = (__bf16*)alloc((size_t)BL*D_MODEL*2);
  __bf16* xbfr    = (__bf16*)alloc((size_t)BL*D_MODEL*2);
  __bf16* winbf   = (__bf16*)alloc((size_t)2*D_IN*D_MODEL*2);
  __bf16* wxbf    = (__bf16*)alloc((size_t)64*D_IN*2);
  __bf16* wdtbf   = (__bf16*)alloc((size_t)D_IN*DT_RANK*2);
  __bf16* woutbf  = (__bf16*)alloc((size_t)D_MODEL*D_IN*2);
  __bf16* wobf    = (__bf16*)alloc((size_t)D_MODEL*(2*D_MODEL)*2);
  float*  xz      = (float*) alloc((size_t)BL*2*D_IN*4);
  float*  xhf     = (float*) alloc((size_t)BL*D_IN*4);
  __bf16* xhbf    = (__bf16*)alloc((size_t)BL*D_IN*2);
  float*  xdbl    = (float*) alloc((size_t)BL*64*4);
  __bf16* dtbf    = (__bf16*)alloc((size_t)BL*DT_RANK*2);
  float*  delta   = (float*) alloc((size_t)BL*D_IN*4);
  float*  ysb     = (float*) alloc((size_t)BL*D_IN*4);
  __bf16* ybf     = (__bf16*)alloc((size_t)BL*D_IN*2);
  float*  outF    = (float*) alloc((size_t)BL*D_MODEL*4);
  float*  outB    = (float*) alloc((size_t)BL*D_MODEL*4);
  __bf16* resbf   = (__bf16*)alloc((size_t)BL*2*D_MODEL*2);
  size_t  stsz    = (size_t)BB*NCHUNK*D_IN*D_STATE*4;
  float*  Pbuf    = (float*) alloc(stsz);
  float*  qbuf    = (float*) alloc(stsz);
  float*  hinbuf  = (float*) alloc(stsz);

  k_lengths<<<dim3(BB), dim3(256), 0, stream>>>(mask, lengths);
  { int n = BL*D_MODEL;
    k_castx<<<dim3((n+255)/256), dim3(256), 0, stream>>>(x, lengths, xbf, xbfr); }
  { int n = D_MODEL*2*D_MODEL;
    k_cast<<<dim3((n+255)/256), dim3(256), 0, stream>>>(Wo, wobf, n); }

  for (int dir=0; dir<2; ++dir){
    const float* const* P = dir ? bP : fP;
    const __bf16* xin = dir ? xbfr : xbf;
    float* outBranch  = dir ? outB : outF;
    { int n=2*D_IN*D_MODEL; k_cast<<<dim3((n+255)/256),dim3(256),0,stream>>>(P[0], winbf, n); }
    { int n=64*D_IN;        k_cast<<<dim3((n+255)/256),dim3(256),0,stream>>>(P[3], wxbf, n); }
    { int n=D_IN*DT_RANK;   k_cast<<<dim3((n+255)/256),dim3(256),0,stream>>>(P[4], wdtbf, n); }
    { int n=D_MODEL*D_IN;   k_cast<<<dim3((n+255)/256),dim3(256),0,stream>>>(P[8], woutbf, n); }
    // xz = x @ W_in^T            (4096 x 2048 x 512)
    launch_gemm(stream, xin, winbf, nullptr, xz, nullptr, BL, 2*D_IN, D_MODEL, 0);
    // depthwise conv + SiLU
    { int n=BL*D_IN; k_conv_silu<<<dim3((n+255)/256),dim3(256),0,stream>>>(xz, P[1], P[2], xhf, xhbf); }
    // x_dbl = xh @ W_x^T         (4096 x 64 x 1024)
    launch_gemm(stream, xhbf, wxbf, nullptr, xdbl, nullptr, BL, 64, D_IN, 0);
    { int n=BL*DT_RANK; k_cast_cols<<<dim3((n+255)/256),dim3(256),0,stream>>>(xdbl, dtbf, BL, 64, DT_RANK); }
    // delta = softplus(dt @ W_dt^T + b_dt)   (4096 x 1024 x 32)
    launch_gemm(stream, dtbf, wdtbf, P[5], delta, nullptr, BL, D_IN, DT_RANK, 1);
    // chunked selective scan
    {
      dim3 blk(128), grd(NCHUNK*D_IN/128, BB);
      k_scan_phase1<<<grd, blk, 0, stream>>>(delta, xhf, xdbl, P[6], Pbuf, qbuf);
      int n2 = BB*D_IN*D_STATE;
      k_scan_phase2<<<dim3((n2+255)/256), dim3(256), 0, stream>>>(Pbuf, qbuf, hinbuf);
      k_scan_phase3<<<grd, blk, 0, stream>>>(delta, xhf, xdbl, P[6], hinbuf, ysb);
    }
    // gate
    { int n=BL*D_IN; k_gate<<<dim3((n+255)/256),dim3(256),0,stream>>>(ysb, xhf, xz, P[7], ybf); }
    // branch out = y @ W_out^T   (4096 x 512 x 1024)
    launch_gemm(stream, ybf, woutbf, nullptr, outBranch, nullptr, BL, D_MODEL, D_IN, 0);
  }
  { int n = BL*2*D_MODEL;
    k_concat<<<dim3((n+255)/256), dim3(256), 0, stream>>>(outF, outB, lengths, resbf); }
  // out = res @ Wo^T + bo        (4096 x 512 x 1024)
  launch_gemm(stream, resbf, wobf, bo, (float*)d_out, nullptr, BL, D_MODEL, 2*D_MODEL, 0);
}
